// NtXentLoss_23081154249195
// MI455X (gfx1250) — matrix operand order repaired and checked by reading, not compiled, not run on hardware
//
#include <hip/hip_runtime.h>

typedef __attribute__((ext_vector_type(16))) _Float16 v16h;
typedef __attribute__((ext_vector_type(8)))  _Float16 v8h;
typedef __attribute__((ext_vector_type(4)))  _Float16 v4h;
typedef __attribute__((ext_vector_type(8)))  float    v8f;

#define NB     4096          // batch per view
#define NTOT   8192          // 2B rows
#define DIM    128           // feature dim
#define INV_T  2.0f          // 1 / 0.5 temperature
#define MSHIFT 2.0f          // fixed logsumexp shift: cos<=1 => sim <= 1/T = 2

#define ROWS_PER_BLOCK 256   // 8 waves x 32 rows
#define STRIPS         16    // column strips
#define STRIP_TILES    32    // 32 tiles x 16 cols = 512 cols per strip
#define LDS_STRIDE     136   // halfs; 272 B pitch => conflict-free ds_load_b128

union V16 { v16h v; v8h h[2]; };

// --------------------------- CDNA5 async staging ---------------------------
#if defined(__has_builtin)
#  if __has_builtin(__builtin_amdgcn_global_load_async_to_lds_b128)
#    define HAVE_ASYNC_LDS 1
#  endif
#endif
#ifndef HAVE_ASYNC_LDS
#  define HAVE_ASYNC_LDS 0
#endif

#if HAVE_ASYNC_LDS
typedef int v4i_ __attribute__((vector_size(16)));              // matches diag type
typedef __attribute__((address_space(1))) v4i_ gas_v4i;         // global (AS1)
typedef __attribute__((address_space(3))) v4i_ las_v4i;         // LDS (AS3)
#endif

__device__ __forceinline__ void stage16(const _Float16* g, _Float16* l) {
#if HAVE_ASYNC_LDS
  __builtin_amdgcn_global_load_async_to_lds_b128(
      (gas_v4i*)g, (las_v4i*)l, 0, 0);
#else
  v8h t = *reinterpret_cast<const v8h*>(g);
  *reinterpret_cast<v8h*>(l) = t;
#endif
}

__device__ __forceinline__ void async_wait0() {
#if HAVE_ASYNC_LDS
#  if __has_builtin(__builtin_amdgcn_s_wait_asynccnt)
  __builtin_amdgcn_s_wait_asynccnt(0);
#  else
  asm volatile("s_wait_asynccnt 0x0" ::: "memory");
#  endif
#endif
}

// ---------------------------------------------------------------------------
// 1) Row-normalize fp32 z -> f16 zn.  One wave per row, 4 floats/lane.
// ---------------------------------------------------------------------------
__global__ __launch_bounds__(256) void nrm_kernel(const float* __restrict__ z1,
                                                  const float* __restrict__ z2,
                                                  _Float16* __restrict__ zn) {
  const int row  = blockIdx.x * 8 + (threadIdx.x >> 5);
  const int lane = threadIdx.x & 31;
  const float* src = (row < NB) ? (z1 + (size_t)row * DIM)
                                : (z2 + (size_t)(row - NB) * DIM);
  float4 v = reinterpret_cast<const float4*>(src)[lane];
  float ss = v.x * v.x + v.y * v.y + v.z * v.z + v.w * v.w;
  #pragma unroll
  for (int off = 16; off > 0; off >>= 1) ss += __shfl_xor(ss, off, 32);
  const float inv = 1.0f / fmaxf(sqrtf(ss), 1e-8f);
  v4h o;
  o[0] = (_Float16)(v.x * inv);
  o[1] = (_Float16)(v.y * inv);
  o[2] = (_Float16)(v.z * inv);
  o[3] = (_Float16)(v.w * inv);
  reinterpret_cast<v4h*>(zn + (size_t)row * DIM)[lane] = o;
}

// ---------------------------------------------------------------------------
// 2) Positive-pair similarity: pos[i] = <zn[i], zn[(i+B)%2B]> / T
// ---------------------------------------------------------------------------
__global__ __launch_bounds__(256) void pos_kernel(const _Float16* __restrict__ zn,
                                                  float* __restrict__ pos) {
  const int row  = blockIdx.x * 8 + (threadIdx.x >> 5);
  const int lane = threadIdx.x & 31;
  const int j    = (row < NB) ? row + NB : row - NB;
  v4h a = reinterpret_cast<const v4h*>(zn + (size_t)row * DIM)[lane];
  v4h b = reinterpret_cast<const v4h*>(zn + (size_t)j   * DIM)[lane];
  float s = (float)a[0] * (float)b[0] + (float)a[1] * (float)b[1]
          + (float)a[2] * (float)b[2] + (float)a[3] * (float)b[3];
  #pragma unroll
  for (int off = 16; off > 0; off >>= 1) s += __shfl_xor(s, off, 32);
  if (lane == 0) pos[row] = s * INV_T;
}

// ---------------------------------------------------------------------------
// 3) Zero per-row exp-sum accumulators (strips combine via atomicAdd).
// ---------------------------------------------------------------------------
__global__ void init_kernel(float* __restrict__ rowsum) {
  const int i = blockIdx.x * 256 + threadIdx.x;
  if (i < NTOT) rowsum[i] = 0.0f;
}

// ---------------------------------------------------------------------------
// 4) Fused GEMM + masked exp-sum.
//    Block = 8 waves; each wave owns TWO 16-row A tiles (32 rows) and reuses
//    each B fragment for two independent WMMA chains (halves DS-load:WMMA
//    ratio, breaks the serial C-dependency).  B tiles double-buffered in LDS;
//    staged with GLOBAL_LOAD_ASYNC_TO_LDS_B128 when available.
//    Diagonal masking hoisted into a wave-uniform branch (C0 == R0/R1).
// ---------------------------------------------------------------------------
__global__ __launch_bounds__(256) void sim_kernel(const _Float16* __restrict__ zn,
                                                  float* __restrict__ rowsum) {
  __shared__ __align__(16) _Float16 bs[2][16 * LDS_STRIDE];

  const int tid  = threadIdx.x;
  const int wave = tid >> 5;
  const int lane = tid & 31;
  const int m    = lane & 15;     // A row / B col / D col within tile
  const int h    = lane >> 4;     // lane half (selects K / M sub-ranges)
  const int R0   = blockIdx.x * ROWS_PER_BLOCK + wave * 32;
  const int R1   = R0 + 16;
  const int C0b  = blockIdx.y * (STRIP_TILES * 16);

  // --- A fragments (ISA 7.12.2, 16-bit A 16x32): lane m holds the row;
  //     elems 0..7 = K 32c+8h.., elems 8..15 = K 32c+16+8h..
  V16 a0[4], a1[4];
  {
    const _Float16* ar0 = zn + (size_t)(R0 + m) * DIM;
    const _Float16* ar1 = zn + (size_t)(R1 + m) * DIM;
    #pragma unroll
    for (int c = 0; c < 4; ++c) {
      a0[c].h[0] = *reinterpret_cast<const v8h*>(ar0 + 32 * c + 8 * h);
      a0[c].h[1] = *reinterpret_cast<const v8h*>(ar0 + 32 * c + 16 + 8 * h);
      a1[c].h[0] = *reinterpret_cast<const v8h*>(ar1 + 32 * c + 8 * h);
      a1[c].h[1] = *reinterpret_cast<const v8h*>(ar1 + 32 * c + 16 + 8 * h);
    }
  }

  // --- cooperative B-tile loader: each of 256 threads moves 16 bytes
  const int lcol = tid >> 4;      // column within tile
  const int lseg = tid & 15;      // 16-byte segment of the column's 128 halfs
  const _Float16* gsrc0 = zn + (size_t)(C0b + lcol) * DIM + lseg * 8;
  _Float16* ldst0 = &bs[0][0] + lcol * LDS_STRIDE + lseg * 8;

  stage16(gsrc0, ldst0);          // preload tile 0 into buffer 0

  float part0[8], part1[8];
  #pragma unroll
  for (int r = 0; r < 8; ++r) { part0[r] = 0.0f; part1[r] = 0.0f; }

  for (int t = 0; t < STRIP_TILES; ++t) {
    async_wait0();                 // our async stores to LDS are done
    __syncthreads();               // everyone's tile is visible
    const int cur = t & 1;

    if (t + 1 < STRIP_TILES) {     // stage next tile into the other buffer
      if (t + 2 < STRIP_TILES)
        __builtin_prefetch(gsrc0 + (size_t)(t + 2) * 16 * DIM, 0, 1);
      stage16(gsrc0 + (size_t)(t + 1) * 16 * DIM,
              ldst0 + (cur ^ 1) * (16 * LDS_STRIDE));
    }

    // --- B fragments (16-bit B 32x16): lane holds col m, elems 0..7 = K 32c+8h.., elems 8..15 = K 32c+16+8h..
    V16 b[4];
    #pragma unroll
    for (int c = 0; c < 4; ++c) {
      const _Float16* p = &bs[cur][m * LDS_STRIDE + 32 * c + 8 * h];
      b[c].h[0] = *reinterpret_cast<const v8h*>(p);
      b[c].h[1] = *reinterpret_cast<const v8h*>(p + 16);
    }

    v8f acc0 = {}, acc1 = {};
    acc0 = __builtin_amdgcn_wmma_f32_16x16x32_f16(false, a0[0].v, false, b[0].v,
                                                  (short)0, acc0, false, false);
    acc1 = __builtin_amdgcn_wmma_f32_16x16x32_f16(false, a1[0].v, false, b[0].v,
                                                  (short)0, acc1, false, false);
    acc0 = __builtin_amdgcn_wmma_f32_16x16x32_f16(false, a0[1].v, false, b[1].v,
                                                  (short)0, acc0, false, false);
    acc1 = __builtin_amdgcn_wmma_f32_16x16x32_f16(false, a1[1].v, false, b[1].v,
                                                  (short)0, acc1, false, false);
    acc0 = __builtin_amdgcn_wmma_f32_16x16x32_f16(false, a0[2].v, false, b[2].v,
                                                  (short)0, acc0, false, false);
    acc1 = __builtin_amdgcn_wmma_f32_16x16x32_f16(false, a1[2].v, false, b[2].v,
                                                  (short)0, acc1, false, false);
    acc0 = __builtin_amdgcn_wmma_f32_16x16x32_f16(false, a0[3].v, false, b[3].v,
                                                  (short)0, acc0, false, false);
    acc1 = __builtin_amdgcn_wmma_f32_16x16x32_f16(false, a1[3].v, false, b[3].v,
                                                  (short)0, acc1, false, false);

    // --- D layout: VGPR r, lane -> row Rx + r + 8h, col C0 + m
    const int C0 = C0b + t * 16;
    const int gc = C0 + m;

    if (C0 == R0) {                // wave-uniform: only tile containing diag
      #pragma unroll
      for (int r = 0; r < 8; ++r) {
        const float e = __expf(acc0[r] * INV_T - MSHIFT);
        part0[r] += ((R0 + r + 8 * h) == gc) ? 0.0f : e;
      }
    } else {
      #pragma unroll
      for (int r = 0; r < 8; ++r)
        part0[r] += __expf(acc0[r] * INV_T - MSHIFT);
    }

    if (C0 == R1) {
      #pragma unroll
      for (int r = 0; r < 8; ++r) {
        const float e = __expf(acc1[r] * INV_T - MSHIFT);
        part1[r] += ((R1 + r + 8 * h) == gc) ? 0.0f : e;
      }
    } else {
      #pragma unroll
      for (int r = 0; r < 8; ++r)
        part1[r] += __expf(acc1[r] * INV_T - MSHIFT);
    }
  }

  // reduce each row's partials across its 16 lanes (xor<16 stays in-half)
  #pragma unroll
  for (int r = 0; r < 8; ++r) {
    float s0 = part0[r];
    float s1 = part1[r];
    s0 += __shfl_xor(s0, 1, 32);  s1 += __shfl_xor(s1, 1, 32);
    s0 += __shfl_xor(s0, 2, 32);  s1 += __shfl_xor(s1, 2, 32);
    s0 += __shfl_xor(s0, 4, 32);  s1 += __shfl_xor(s1, 4, 32);
    s0 += __shfl_xor(s0, 8, 32);  s1 += __shfl_xor(s1, 8, 32);
    if (m == 0) {
      atomicAdd(&rowsum[R0 + r + 8 * h], s0);
      atomicAdd(&rowsum[R1 + r + 8 * h], s1);
    }
  }
}

// ---------------------------------------------------------------------------
// 5) loss = mean_i( 2 + log(rowsum_i) - pos_i )
// ---------------------------------------------------------------------------
__global__ __launch_bounds__(256) void loss_kernel(const float* __restrict__ rowsum,
                                                   const float* __restrict__ pos,
                                                   float* __restrict__ out) {
  __shared__ float red[256];
  float acc = 0.0f;
  for (int i = threadIdx.x; i < NTOT; i += 256)
    acc += (MSHIFT + __logf(rowsum[i]) - pos[i]);
  red[threadIdx.x] = acc;
  __syncthreads();
  for (int s = 128; s > 0; s >>= 1) {
    if (threadIdx.x < s) red[threadIdx.x] += red[threadIdx.x + s];
    __syncthreads();
  }
  if (threadIdx.x == 0) out[0] = red[0] / (float)NTOT;
}

// ---------------------------------------------------------------------------
extern "C" void kernel_launch(void* const* d_in, const int* in_sizes, int n_in,
                              void* d_out, int out_size, void* d_ws, size_t ws_size,
                              hipStream_t stream) {
  (void)in_sizes; (void)n_in; (void)out_size; (void)ws_size;
  const float* z1 = (const float*)d_in[0];
  const float* z2 = (const float*)d_in[1];

  char* ws = (char*)d_ws;
  _Float16* zn  = (_Float16*)ws;                               // 2 MB
  float* rowsum = (float*)(ws + (size_t)NTOT * DIM * 2);       // 32 KB
  float* pos    = rowsum + NTOT;                               // 32 KB
  float* out    = (float*)d_out;

  nrm_kernel <<<NTOT / 8,   256, 0, stream>>>(z1, z2, zn);
  pos_kernel <<<NTOT / 8,   256, 0, stream>>>(zn, pos);
  init_kernel<<<NTOT / 256, 256, 0, stream>>>(rowsum);
  sim_kernel <<<dim3(NTOT / ROWS_PER_BLOCK, STRIPS), 256, 0, stream>>>(zn, rowsum);
  loss_kernel<<<1,          256, 0, stream>>>(rowsum, pos, out);
}
